// SwinTransformerBlock_31860067402155
// MI455X (gfx1250) — matrix operand order repaired and checked by reading, not compiled, not run on hardware
//
#include <hip/hip_runtime.h>
#include <hip/hip_bf16.h>
#include <math.h>

typedef __attribute__((ext_vector_type(16))) _Float16 v16h;
typedef __attribute__((ext_vector_type(8)))  float    v8f;

#define BB      32
#define HH      56
#define WW      56
#define CC      192
#define HEADS   6
#define WSZ     7
#define SHIFT   3
#define NTOK    49
#define HDIM    32
#define HIDDEN  768
#define NWIN    2048            // BB * 64 windows
#define MROWS   (NWIN * NTOK)   // 100352
#define LDST    48              // padded LDS K-stride (f16 elems): 96B, 16B-aligned chunks
#define A_TILE  (128 * LDST)    // f16 elems per A buffer
#define B_TILE  (64 * LDST)     // f16 elems per B buffer

static __device__ __forceinline__ v8f vzero() {
  v8f z = {0.f, 0.f, 0.f, 0.f, 0.f, 0.f, 0.f, 0.f};
  return z;
}

// A-fragment (16x32 f16, MxK): lane = {half, m}; contiguous k order:
// pair p -> k = half*16 + 2*p
static __device__ __forceinline__ v16h frag_ldA(const _Float16* base, int row0, int k0, int ld) {
  const int lane = threadIdx.x & 31;
  const int m = lane & 15, hf = lane >> 4;
  v16h f;
#pragma unroll
  for (int p = 0; p < 8; ++p) {
    int k = k0 + hf * 16 + (p << 1);
    const _Float16* s = base + (row0 + m) * ld + k;
    f[2 * p]     = s[0];
    f[2 * p + 1] = s[1];
  }
  return f;
}

// B-fragment (32x16 f16, KxN) from an [N][K] row-major array (i.e. B = stored^T):
// contiguous k order, as in frag_ldA: lanes 0-15 hold K=0..15, lanes 16-31 hold K=16..31
static __device__ __forceinline__ v16h frag_ldB(const _Float16* base, int row0, int k0, int ld) {
  const int lane = threadIdx.x & 31;
  const int n = lane & 15, hf = lane >> 4;
  v16h f;
#pragma unroll
  for (int p = 0; p < 8; ++p) {
    int k = k0 + hf * 16 + (p << 1);
    const _Float16* s = base + (row0 + n) * ld + k;
    f[2 * p]     = s[0];
    f[2 * p + 1] = s[1];
  }
  return f;
}

// B-fragment from a [K][N] row-major array (true KxN storage, e.g. V in attention)
static __device__ __forceinline__ v16h frag_ldBT(const _Float16* base, int n0, int k0, int ld) {
  const int lane = threadIdx.x & 31;
  const int n = (lane & 15) + n0, hf = lane >> 4;
  v16h f;
#pragma unroll
  for (int p = 0; p < 8; ++p) {
    int k = k0 + hf * 16 + (p << 1);
    f[2 * p]     = base[k * ld + n];
    f[2 * p + 1] = base[(k + 1) * ld + n];
  }
  return f;
}

static __device__ __forceinline__ v8f wmma16(v16h a, v16h b, v8f c) {
  return __builtin_amdgcn_wmma_f32_16x16x32_f16(false, a, false, b, (short)0, c, false, false);
}

// CDNA5 async copy: global -> LDS, 16B per lane, tracked by ASYNCcnt.
static __device__ __forceinline__ void async_ld16(unsigned lds_off, const _Float16* gptr) {
  asm volatile("global_load_async_to_lds_b128 %0, %1, off"
               :: "v"(lds_off), "v"((unsigned long)(size_t)gptr)
               : "memory");
}
static __device__ __forceinline__ void wait_async_le3() {
  asm volatile("s_wait_asynccnt 0x3" ::: "memory");
}
static __device__ __forceinline__ void wait_async_0() {
  asm volatile("s_wait_asynccnt 0x0" ::: "memory");
}

// ---- 128x64 WMMA GEMM core, double-buffered LDS fed by async global->LDS DMA ----
// C = A[M,K] * W[N,K]^T.  sA: 2*A_TILE f16, sB: 2*B_TILE f16.
static __device__ __forceinline__ void gemm_core(const _Float16* __restrict__ A,
                                                 const _Float16* __restrict__ Wt,
                                                 int K, _Float16* sA, _Float16* sB,
                                                 v8f acc[2][2]) {
  const int tid = threadIdx.x;
  const long m0 = (long)blockIdx.x * 128;
  const long n0 = (long)blockIdx.y * 64;
  acc[0][0] = vzero(); acc[0][1] = vzero(); acc[1][0] = vzero(); acc[1][1] = vzero();
  const int wv = tid >> 5, wm = wv & 3, wn = wv >> 2;

  // per-thread chunk coordinates (8 f16 = 16B per async op)
  int arow[2], acol[2];
#pragma unroll
  for (int c = 0; c < 2; ++c) {
    int chunk = tid * 2 + c;              // A tile: 512 chunks of 8 f16
    arow[c] = chunk >> 2;
    acol[c] = (chunk & 3) << 3;
  }
  const int brow = tid >> 2;              // B tile: 256 chunks
  const int bcol = (tid & 3) << 3;
  unsigned la[2], lb;
#pragma unroll
  for (int c = 0; c < 2; ++c)
    la[c] = (unsigned)(size_t)(sA + arow[c] * LDST + acol[c]);
  lb = (unsigned)(size_t)(sB + brow * LDST + bcol);
  const _Float16* gA0 = A + (m0 + arow[0]) * K + acol[0];
  const _Float16* gA1 = A + (m0 + arow[1]) * K + acol[1];
  const _Float16* gB  = Wt + (n0 + brow) * K + bcol;

  const int NT = K >> 5;
  // prologue: stage k-tile 0 into buffer 0
  async_ld16(la[0], gA0);
  async_ld16(la[1], gA1);
  async_ld16(lb, gB);

  for (int kt = 0; kt < NT; ++kt) {
    const int cur = kt & 1;
    if (kt + 1 < NT) {                    // prefetch next tile into spare buffer
      const int k1 = (kt + 1) << 5;
      const unsigned abuf = (cur ^ 1) * (A_TILE * 2);   // byte stride of A buffer
      const unsigned bbuf = (cur ^ 1) * (B_TILE * 2);
      async_ld16(la[0] + abuf, gA0 + k1);
      async_ld16(la[1] + abuf, gA1 + k1);
      async_ld16(lb + bbuf, gB + k1);
      wait_async_le3();                   // older 3 (current tile) are done; next 3 in flight
    } else {
      wait_async_0();
    }
    __syncthreads();                      // current buffer visible to all waves
    const _Float16* cA = sA + cur * A_TILE;
    const _Float16* cB = sB + cur * B_TILE;
    v16h a0 = frag_ldA(cA, wm * 32,      0, LDST);
    v16h a1 = frag_ldA(cA, wm * 32 + 16, 0, LDST);
    v16h b0 = frag_ldB(cB, wn * 32,      0, LDST);
    v16h b1 = frag_ldB(cB, wn * 32 + 16, 0, LDST);
    acc[0][0] = wmma16(a0, b0, acc[0][0]);
    acc[0][1] = wmma16(a0, b1, acc[0][1]);
    acc[1][0] = wmma16(a1, b0, acc[1][0]);
    acc[1][1] = wmma16(a1, b1, acc[1][1]);
    __syncthreads();                      // reads done before next iter overwrites this buffer
  }
}

// ---------------- GEMM kernels with fused epilogues ----------------
__global__ __launch_bounds__(256)
void gemm_qkv(const _Float16* __restrict__ A, const _Float16* __restrict__ Wt,
              const float* __restrict__ qb, const float* __restrict__ vb,
              _Float16* __restrict__ out) {
  __shared__ _Float16 sA[2 * A_TILE];
  __shared__ _Float16 sB[2 * B_TILE];
  v8f acc[2][2];
  gemm_core(A, Wt, CC, sA, sB, acc);
  const int tid = threadIdx.x, lane = tid & 31, wv = tid >> 5, wm = wv & 3, wn = wv >> 2;
#pragma unroll
  for (int i = 0; i < 2; ++i)
#pragma unroll
    for (int j = 0; j < 2; ++j)
#pragma unroll
      for (int r = 0; r < 8; ++r) {
        long m = (long)blockIdx.x * 128 + wm * 32 + i * 16 + (lane >> 4) * 8 + r;
        int n = blockIdx.y * 64 + wn * 32 + j * 16 + (lane & 15);
        float bias = (n < CC) ? qb[n] : ((n < 2 * CC) ? 0.f : vb[n - 2 * CC]);
        out[m * (3 * CC) + n] = (_Float16)(acc[i][j][r] + bias);
      }
}

__global__ __launch_bounds__(256)
void gemm_proj(const _Float16* __restrict__ A, const _Float16* __restrict__ Wt,
               const float* __restrict__ pb, float* __restrict__ tmp) {
  __shared__ _Float16 sA[2 * A_TILE];
  __shared__ _Float16 sB[2 * B_TILE];
  v8f acc[2][2];
  gemm_core(A, Wt, CC, sA, sB, acc);
  const int tid = threadIdx.x, lane = tid & 31, wv = tid >> 5, wm = wv & 3, wn = wv >> 2;
#pragma unroll
  for (int i = 0; i < 2; ++i)
#pragma unroll
    for (int j = 0; j < 2; ++j)
#pragma unroll
      for (int r = 0; r < 8; ++r) {
        int m = blockIdx.x * 128 + wm * 32 + i * 16 + (lane >> 4) * 8 + r;
        int n = blockIdx.y * 64 + wn * 32 + j * 16 + (lane & 15);
        // window row -> image row with reverse roll (+SHIFT)
        int win = m / NTOK, t = m - win * NTOK;
        int b = win >> 6, w64 = win & 63;
        int wh = w64 >> 3, ww = w64 & 7;
        int rr = t / WSZ, cc = t - rr * WSZ;
        int h = wh * WSZ + rr + SHIFT; if (h >= HH) h -= HH;
        int w = ww * WSZ + cc + SHIFT; if (w >= WW) w -= WW;
        long irow = (long)b * (HH * WW) + h * WW + w;
        tmp[irow * CC + n] = acc[i][j][r] + pb[n];
      }
}

__global__ __launch_bounds__(256)
void gemm_ff1(const _Float16* __restrict__ A, const _Float16* __restrict__ Wt,
              const float* __restrict__ b1, _Float16* __restrict__ out) {
  __shared__ _Float16 sA[2 * A_TILE];
  __shared__ _Float16 sB[2 * B_TILE];
  v8f acc[2][2];
  gemm_core(A, Wt, CC, sA, sB, acc);
  const int tid = threadIdx.x, lane = tid & 31, wv = tid >> 5, wm = wv & 3, wn = wv >> 2;
#pragma unroll
  for (int i = 0; i < 2; ++i)
#pragma unroll
    for (int j = 0; j < 2; ++j)
#pragma unroll
      for (int r = 0; r < 8; ++r) {
        long m = (long)blockIdx.x * 128 + wm * 32 + i * 16 + (lane >> 4) * 8 + r;
        int n = blockIdx.y * 64 + wn * 32 + j * 16 + (lane & 15);
        float t = acc[i][j][r] + b1[n];
        float g = 0.5f * t * (1.f + erff(t * 0.70710678118f));  // exact GELU
        out[m * HIDDEN + n] = (_Float16)g;
      }
}

__global__ __launch_bounds__(256)
void gemm_ff2(const _Float16* __restrict__ A, const _Float16* __restrict__ Wt,
              const float* __restrict__ b2, float* __restrict__ tmp) {
  __shared__ _Float16 sA[2 * A_TILE];
  __shared__ _Float16 sB[2 * B_TILE];
  v8f acc[2][2];
  gemm_core(A, Wt, HIDDEN, sA, sB, acc);
  const int tid = threadIdx.x, lane = tid & 31, wv = tid >> 5, wm = wv & 3, wn = wv >> 2;
#pragma unroll
  for (int i = 0; i < 2; ++i)
#pragma unroll
    for (int j = 0; j < 2; ++j)
#pragma unroll
      for (int r = 0; r < 8; ++r) {
        long m = (long)blockIdx.x * 128 + wm * 32 + i * 16 + (lane >> 4) * 8 + r;
        int n = blockIdx.y * 64 + wn * 32 + j * 16 + (lane & 15);
        tmp[m * CC + n] = acc[i][j][r] + b2[n];
      }
}

// ---------------- attention: one block per (window, head) ----------------
__global__ __launch_bounds__(128)
void attn_kernel(const _Float16* __restrict__ qkv, const float* __restrict__ tbl,
                 const float* __restrict__ logit_scale, _Float16* __restrict__ attn_out) {
  __shared__ _Float16 sQ[64 * 36];
  __shared__ _Float16 sK[64 * 36];
  __shared__ _Float16 sV[64 * 36];
  __shared__ float    sS[64 * 65];
  __shared__ _Float16 sP[64 * 68];
  const int tid = threadIdx.x;
  const int blk = blockIdx.x;
  const int win = blk / HEADS;
  const int head = blk - win * HEADS;
  const int qo = head * HDIM, ko = CC + head * HDIM, vo = 2 * CC + head * HDIM;
  const long rowbase = (long)win * NTOK;

  for (int e = tid; e < 64 * 32; e += 128) {            // load q,k,v (rows>=49 zero)
    int r = e >> 5, c = e & 31;
    _Float16 q = (_Float16)0.f, k = (_Float16)0.f, v = (_Float16)0.f;
    if (r < NTOK) {
      const _Float16* src = qkv + (rowbase + r) * (3 * CC);
      q = src[qo + c]; k = src[ko + c]; v = src[vo + c];
    }
    sQ[r * 36 + c] = q; sK[r * 36 + c] = k; sV[r * 36 + c] = v;
  }
  __syncthreads();
  {                                                      // l2-normalize q,k rows
    _Float16* mat = (tid < 64) ? sQ : sK;
    int r = tid & 63;
    float ssq = 0.f;
    for (int c = 0; c < 32; ++c) { float x = (float)mat[r * 36 + c]; ssq += x * x; }
    float inv = 1.f / fmaxf(sqrtf(ssq), 1e-12f);
    for (int c = 0; c < 32; ++c) mat[r * 36 + c] = (_Float16)((float)mat[r * 36 + c] * inv);
  }
  __syncthreads();
  const int wv = tid >> 5, lane = tid & 31;
  const float lscale = expf(fminf(logit_scale[head], 4.60517019f));  // ln(100)
  const int w64 = win & 63, wh = w64 >> 3, ww = w64 & 7;
  {                                                      // S = q * k^T (cosine attn)
    v16h aq = frag_ldA(sQ, 16 * wv, 0, 36);
#pragma unroll
    for (int jb = 0; jb < 4; ++jb) {
      v16h bk = frag_ldB(sK, 16 * jb, 0, 36);
      v8f sacc = vzero();
      sacc = wmma16(aq, bk, sacc);
#pragma unroll
      for (int r = 0; r < 8; ++r) {
        int i = 16 * wv + (lane >> 4) * 8 + r;
        int j = 16 * jb + (lane & 15);
        float val;
        if (i < NTOK && j < NTOK) {
          int iy = i / WSZ, ix = i - iy * WSZ, jy = j / WSZ, jx = j - jy * WSZ;
          float raw = tbl[((iy - jy + 6) * 13 + (ix - jx + 6)) * HEADS + head];
          float rpb = 16.f / (1.f + expf(-raw));         // 16 * sigmoid
          int gih = wh * WSZ + iy, giw = ww * WSZ + ix;
          int gjh = wh * WSZ + jy, gjw = ww * WSZ + jx;
          int ri = (gih < 49 ? 0 : (gih < 53 ? 1 : 2)) * 3 + (giw < 49 ? 0 : (giw < 53 ? 1 : 2));
          int rj = (gjh < 49 ? 0 : (gjh < 53 ? 1 : 2)) * 3 + (gjw < 49 ? 0 : (gjw < 53 ? 1 : 2));
          val = sacc[r] * lscale + rpb + ((ri == rj) ? 0.f : -100.f);
        } else {
          val = (j >= NTOK) ? -1e30f : 0.f;
        }
        sS[i * 65 + j] = val;
      }
    }
  }
  __syncthreads();
  if (tid < 64) {                                        // softmax over 49 cols
    int r = tid;
    float mx = -1e30f;
    for (int j = 0; j < NTOK; ++j) mx = fmaxf(mx, sS[r * 65 + j]);
    float sum = 0.f;
    for (int j = 0; j < NTOK; ++j) { float e = expf(sS[r * 65 + j] - mx); sS[r * 65 + j] = e; sum += e; }
    float inv = 1.f / sum;
    for (int j = 0; j < NTOK; ++j) sP[r * 68 + j] = (_Float16)(sS[r * 65 + j] * inv);
    for (int j = NTOK; j < 64; ++j) sP[r * 68 + j] = (_Float16)0.f;
  }
  __syncthreads();
  {                                                      // O = P * V
    v8f o0 = vzero(), o1 = vzero();
#pragma unroll
    for (int kb = 0; kb < 2; ++kb) {
      v16h ap = frag_ldA(sP, 16 * wv, 32 * kb, 68);
      v16h bv0 = frag_ldBT(sV, 0, 32 * kb, 36);
      v16h bv1 = frag_ldBT(sV, 16, 32 * kb, 36);
      o0 = wmma16(ap, bv0, o0);
      o1 = wmma16(ap, bv1, o1);
    }
#pragma unroll
    for (int r = 0; r < 8; ++r) {
      int i = 16 * wv + (lane >> 4) * 8 + r;
      if (i < NTOK) {
        int n = lane & 15;
        _Float16* dst = attn_out + (rowbase + i) * CC + head * HDIM;
        dst[n]      = (_Float16)o0[r];
        dst[16 + n] = (_Float16)o1[r];
      }
    }
  }
}

// ---------------- small kernels ----------------
__global__ void cvt_kernel(const float* __restrict__ s, _Float16* __restrict__ d, int n) {
  int i = blockIdx.x * 256 + threadIdx.x;
  if (i < n) d[i] = (_Float16)s[i];
}

static __device__ __forceinline__ float cpb_coord(int i) {
  float v = (float)(i - 6) * (8.f / 6.f);
  float m = log2f(fabsf(v) + 1.f) * (1.f / 3.f);   // /log2(8)
  return (v < 0.f) ? -m : m;
}

__global__ __launch_bounds__(512)
void cpb_kernel(const float* __restrict__ w1, const float* __restrict__ b1,
                const float* __restrict__ w2, float* __restrict__ tbl) {
  __shared__ float red[512];
  int e = blockIdx.x, j = threadIdx.x;
  int ih = e / 13, iw = e - ih * 13;
  float f0 = cpb_coord(ih), f1 = cpb_coord(iw);
  float h = fmaxf(0.f, w1[2 * j] * f0 + w1[2 * j + 1] * f1 + b1[j]);
  for (int hd = 0; hd < HEADS; ++hd) {
    red[j] = h * w2[hd * 512 + j];
    __syncthreads();
    for (int st = 256; st > 0; st >>= 1) {
      if (j < st) red[j] += red[j + st];
      __syncthreads();
    }
    if (j == 0) tbl[e * HEADS + hd] = red[0];
    __syncthreads();
  }
}

__global__ void gather_kernel(const float* __restrict__ x, _Float16* __restrict__ xw) {
  int g = blockIdx.x * 256 + threadIdx.x;          // < 100352*192 fits in int
  int m = g / CC, ch = g - m * CC;
  int win = m / NTOK, t = m - win * NTOK;
  int b = win >> 6, w64 = win & 63;
  int wh = w64 >> 3, ww = w64 & 7;
  int r = t / WSZ, c = t - r * WSZ;
  int hs = wh * WSZ + r + SHIFT; if (hs >= HH) hs -= HH;   // roll(-SHIFT)
  int wsd = ww * WSZ + c + SHIFT; if (wsd >= WW) wsd -= WW;
  xw[g] = (_Float16)x[((long)b * (HH * WW) + hs * WW + wsd) * CC + ch];
}

__global__ __launch_bounds__(256)
void ln_kernel(const float* __restrict__ shortcut, const float* __restrict__ y,
               const float* __restrict__ g, const float* __restrict__ bta,
               float* __restrict__ out, _Float16* __restrict__ out_h) {
  int row = blockIdx.x * 8 + (threadIdx.x >> 5);   // one wave32 per token row
  int lane = threadIdx.x & 31;
  const float* yr = y + (long)row * CC;
  float v[6];
  float s = 0.f, sq = 0.f;
#pragma unroll
  for (int t = 0; t < 6; ++t) { float x = yr[lane + 32 * t]; v[t] = x; s += x; sq += x * x; }
#pragma unroll
  for (int m = 16; m >= 1; m >>= 1) { s += __shfl_xor(s, m, 32); sq += __shfl_xor(sq, m, 32); }
  float mean = s * (1.f / CC);
  float var = sq * (1.f / CC) - mean * mean;
  float inv = rsqrtf(var + 1e-5f);
  const float* sc = shortcut + (long)row * CC;
  float* orow = out + (long)row * CC;
#pragma unroll
  for (int t = 0; t < 6; ++t) {
    int ch = lane + 32 * t;
    float o = sc[ch] + (v[t] - mean) * inv * g[ch] + bta[ch];
    orow[ch] = o;
    if (out_h) out_h[(long)row * CC + ch] = (_Float16)o;
  }
}

// ---------------- host launcher ----------------
extern "C" void kernel_launch(void* const* d_in, const int* in_sizes, int n_in,
                              void* d_out, int out_size, void* d_ws, size_t ws_size,
                              hipStream_t stream) {
  const float* x           = (const float*)d_in[0];
  const float* qkv_w       = (const float*)d_in[1];
  const float* q_bias      = (const float*)d_in[2];
  const float* v_bias      = (const float*)d_in[3];
  const float* logit_scale = (const float*)d_in[4];
  const float* cpb_w1      = (const float*)d_in[5];
  const float* cpb_b1      = (const float*)d_in[6];
  const float* cpb_w2      = (const float*)d_in[7];
  const float* proj_w      = (const float*)d_in[8];
  const float* proj_b      = (const float*)d_in[9];
  const float* n1g         = (const float*)d_in[10];
  const float* n1b         = (const float*)d_in[11];
  const float* n2g         = (const float*)d_in[12];
  const float* n2b         = (const float*)d_in[13];
  const float* ff_w1       = (const float*)d_in[14];
  const float* ff_b1       = (const float*)d_in[15];
  const float* ff_w2       = (const float*)d_in[16];
  const float* ff_b2       = (const float*)d_in[17];
  float* out = (float*)d_out;

  char* ws = (char*)d_ws;
  size_t off = 0;
  auto take = [&](size_t bytes) -> char* {
    char* p = ws + off;
    off = (off + bytes + 255) & ~(size_t)255;
    return p;
  };
  _Float16* wqkv_h  = (_Float16*)take((size_t)3 * CC * CC * 2);
  _Float16* wproj_h = (_Float16*)take((size_t)CC * CC * 2);
  _Float16* wff1_h  = (_Float16*)take((size_t)HIDDEN * CC * 2);
  _Float16* wff2_h  = (_Float16*)take((size_t)CC * HIDDEN * 2);
  float*    tbl     = (float*)take((size_t)169 * HEADS * 4);
  _Float16* xw_h    = (_Float16*)take((size_t)MROWS * CC * 2);
  _Float16* qkv_h   = (_Float16*)take((size_t)MROWS * 3 * CC * 2);
  _Float16* attn_h  = (_Float16*)take((size_t)MROWS * CC * 2);
  float*    tmp     = (float*)take((size_t)MROWS * CC * 4);
  float*    x1      = (float*)take((size_t)MROWS * CC * 4);
  _Float16* x1_h    = (_Float16*)take((size_t)MROWS * CC * 2);
  _Float16* ff1_h   = qkv_h;  // reuse dead qkv+attn regions (exactly 100352*768 f16)

  cvt_kernel<<<(3 * CC * CC + 255) / 256, 256, 0, stream>>>(qkv_w, wqkv_h, 3 * CC * CC);
  cvt_kernel<<<(CC * CC + 255) / 256, 256, 0, stream>>>(proj_w, wproj_h, CC * CC);
  cvt_kernel<<<(HIDDEN * CC + 255) / 256, 256, 0, stream>>>(ff_w1, wff1_h, HIDDEN * CC);
  cvt_kernel<<<(CC * HIDDEN + 255) / 256, 256, 0, stream>>>(ff_w2, wff2_h, CC * HIDDEN);
  cpb_kernel<<<169, 512, 0, stream>>>(cpb_w1, cpb_b1, cpb_w2, tbl);
  gather_kernel<<<(MROWS * CC) / 256, 256, 0, stream>>>(x, xw_h);
  gemm_qkv<<<dim3(MROWS / 128, (3 * CC) / 64), 256, 0, stream>>>(xw_h, wqkv_h, q_bias, v_bias, qkv_h);
  attn_kernel<<<NWIN * HEADS, 128, 0, stream>>>(qkv_h, tbl, logit_scale, attn_h);
  gemm_proj<<<dim3(MROWS / 128, CC / 64), 256, 0, stream>>>(attn_h, wproj_h, proj_b, tmp);
  ln_kernel<<<MROWS / 8, 256, 0, stream>>>(x, tmp, n1g, n1b, x1, x1_h);
  gemm_ff1<<<dim3(MROWS / 128, HIDDEN / 64), 256, 0, stream>>>(x1_h, wff1_h, ff_b1, ff1_h);
  gemm_ff2<<<dim3(MROWS / 128, CC / 64), 256, 0, stream>>>(ff1_h, wff2_h, ff_b2, tmp);
  ln_kernel<<<MROWS / 8, 256, 0, stream>>>(x1, tmp, n2g, n2b, out, (_Float16*)nullptr);
}
